// Calcium_Net_57827439673704
// MI455X (gfx1250) — matrix operand order repaired and checked by reading, not compiled, not run on hardware
//
#include <hip/hip_runtime.h>

typedef __attribute__((ext_vector_type(16))) _Float16 v16h;
typedef __attribute__((ext_vector_type(8)))  _Float16 v8h;
typedef __attribute__((ext_vector_type(8)))  float    v8f;
typedef __attribute__((ext_vector_type(4)))  float    v4f_;

#define TT 400
#define BB 128
#define NN 256
#define EE 256
#define CC 256
#define UU 128
#define FF 64
#define GG 192           // U + F
#define G3E 768          // 3*E == 3*C
#define CLIPV 5.0f
#define DAMPF 0.9f

// ---------------------------------------------------------------------------
// Packed-B layout: for column-tile ct and k-chunk kc (32 K-values), a block of
// 512 f16: lane L (0..31) holds 16 halves at offset ((ct*KC+kc)*32 + L)*16,
// contents B[k][n] = W[n][k] with n = ct*16+(L&15), k = kc*32 + 16*(j>>3) + 8*(L>>4) + (j&7)
// (the ISA 16-bit B 32x16 fragment). One 32-byte load per WMMA.
// ---------------------------------------------------------------------------
__global__ void __launch_bounds__(512)
pack_weight(const float* __restrict__ W, int ldw, int co, int K, int Nc,
            _Float16* __restrict__ out) {
  const int i = blockIdx.x * blockDim.x + threadIdx.x;
  if (i >= Nc * K) return;
  const int KC = K >> 5;
  const int j = i & 15;
  const int lane = (i >> 4) & 31;
  const int blk = i >> 9;
  const int kc = blk % KC;
  const int ct = blk / KC;
  const int n = (ct << 4) + (lane & 15);
  const int k = (kc << 5) + ((j >> 3) << 4) + ((lane >> 4) << 3) + (j & 7);
  out[i] = (_Float16)W[(size_t)n * ldw + co + k];
}

// ---------------------------------------------------------------------------
// A-fragment loaders (ISA 16-bit A 16x32 layout):
// lane m = L&15, hi = L>>4; halves 0..7 -> K = k0+8*hi+j ; 8..15 -> +16
// ---------------------------------------------------------------------------
__device__ __forceinline__ v16h ldA_f32(const float* __restrict__ A, int lda,
                                        int row0, int k0) {
  const int lane = threadIdx.x & 31;
  const int m = lane & 15, hi = lane >> 4;
  const float* p = A + (size_t)(row0 + m) * lda + (k0 + 8 * hi);
  const v4f_ a0 = *(const v4f_*)(p);
  const v4f_ a1 = *(const v4f_*)(p + 4);
  const v4f_ a2 = *(const v4f_*)(p + 16);
  const v4f_ a3 = *(const v4f_*)(p + 20);
  v16h a;
#pragma unroll
  for (int j = 0; j < 4; ++j) {
    a[j]      = (_Float16)a0[j];
    a[4 + j]  = (_Float16)a1[j];
    a[8 + j]  = (_Float16)a2[j];
    a[12 + j] = (_Float16)a3[j];
  }
  return a;
}

__device__ __forceinline__ v16h ldA_h16(const _Float16* __restrict__ A, int lda,
                                        int row0, int k0) {
  const int lane = threadIdx.x & 31;
  const int m = lane & 15, hi = lane >> 4;
  const _Float16* p = A + (size_t)(row0 + m) * lda + (k0 + 8 * hi);
  const v8h lo = *(const v8h*)(p);
  const v8h hh = *(const v8h*)(p + 16);
  v16h a;
#pragma unroll
  for (int j = 0; j < 8; ++j) { a[j] = lo[j]; a[8 + j] = hh[j]; }
  return a;
}

// ---------------------------------------------------------------------------
// 4-wide K-loop cores: one A-fragment feeds 4 adjacent column tiles.
// KC is compile-time -> full unroll, cross-iteration load pipelining.
// B fragments hoisted before the 4 WMMAs -> staged loadcnt waits.
// ---------------------------------------------------------------------------
template <int KC>
__device__ __forceinline__ void wmma4_f32A(const float* __restrict__ A, int lda,
                                           int row0,
                                           const _Float16* __restrict__ Wp,
                                           int ct0, v8f acc[4]) {
  const int lane = threadIdx.x & 31;
  const size_t cstride = (size_t)KC << 9;
  const _Float16* bp = Wp + (size_t)ct0 * cstride + (lane << 4);
#pragma unroll
  for (int kc = 0; kc < KC; ++kc) {
    const v16h b0 = *(const v16h*)(bp);
    const v16h b1 = *(const v16h*)(bp + cstride);
    const v16h b2 = *(const v16h*)(bp + 2 * cstride);
    const v16h b3 = *(const v16h*)(bp + 3 * cstride);
    const v16h a = ldA_f32(A, lda, row0, kc << 5);
    acc[0] = __builtin_amdgcn_wmma_f32_16x16x32_f16(false, a, false, b0,
                                                    (short)0, acc[0], false, false);
    acc[1] = __builtin_amdgcn_wmma_f32_16x16x32_f16(false, a, false, b1,
                                                    (short)0, acc[1], false, false);
    acc[2] = __builtin_amdgcn_wmma_f32_16x16x32_f16(false, a, false, b2,
                                                    (short)0, acc[2], false, false);
    acc[3] = __builtin_amdgcn_wmma_f32_16x16x32_f16(false, a, false, b3,
                                                    (short)0, acc[3], false, false);
    bp += 512;
  }
}

template <int KC>
__device__ __forceinline__ void wmma4_h16A(const _Float16* __restrict__ A, int lda,
                                           int row0,
                                           const _Float16* __restrict__ Wp,
                                           int ct0, v8f acc[4]) {
  const int lane = threadIdx.x & 31;
  const size_t cstride = (size_t)KC << 9;
  const _Float16* bp = Wp + (size_t)ct0 * cstride + (lane << 4);
#pragma unroll
  for (int kc = 0; kc < KC; ++kc) {
    const v16h b0 = *(const v16h*)(bp);
    const v16h b1 = *(const v16h*)(bp + cstride);
    const v16h b2 = *(const v16h*)(bp + 2 * cstride);
    const v16h b3 = *(const v16h*)(bp + 3 * cstride);
    const v16h a = ldA_h16(A, lda, row0, kc << 5);
    acc[0] = __builtin_amdgcn_wmma_f32_16x16x32_f16(false, a, false, b0,
                                                    (short)0, acc[0], false, false);
    acc[1] = __builtin_amdgcn_wmma_f32_16x16x32_f16(false, a, false, b1,
                                                    (short)0, acc[1], false, false);
    acc[2] = __builtin_amdgcn_wmma_f32_16x16x32_f16(false, a, false, b2,
                                                    (short)0, acc[2], false, false);
    acc[3] = __builtin_amdgcn_wmma_f32_16x16x32_f16(false, a, false, b3,
                                                    (short)0, acc[3], false, false);
    bp += 512;
  }
}

// C/D layout: lane n = L&15, hi = L>>4; VGPR r -> M = row0 + r + 8*hi
__device__ __forceinline__ void st_tile(float* __restrict__ Out, int ldo,
                                        int row0, int col0, v8f acc) {
  const int lane = threadIdx.x & 31;
  const int n = col0 + (lane & 15), hi = lane >> 4;
  float* p = Out + (size_t)(row0 + 8 * hi) * ldo + n;
#pragma unroll
  for (int r = 0; r < 8; ++r) p[(size_t)r * ldo] = acc[r];
}

__device__ __forceinline__ float sigmoidf_(float x) {
  return 1.0f / (1.0f + __expf(-x));
}

// prefetch a whole row-major slab (one prefetch per 64B line)
__device__ __forceinline__ void prefetch_slab(const float* __restrict__ p, int n) {
  for (int e = threadIdx.x << 4; e < n; e += blockDim.x << 4)
    __builtin_prefetch(p + e, 0, 1);
}

// ---------------------------------------------------------------------------
// Phase 1 / 3: big parallel GEMM  Out[M x Nc] = A @ W^T + bias (W packed f16)
// one wave computes a 16x64 strip (4 column tiles) of the output
// ---------------------------------------------------------------------------
template <int KC>
__global__ void __launch_bounds__(256)
gemm_wmma(const float* __restrict__ A, int lda,
          const _Float16* __restrict__ Wp,
          const float* __restrict__ bias,
          float* __restrict__ Out, int ldo, int M, int Ncols) {
  const int grpN = Ncols >> 6;                    // groups of 4 column tiles
  const int units = (M >> 4) * grpN;
  const int wave = (blockIdx.x * blockDim.x + threadIdx.x) >> 5;
  if (wave >= units) return;                      // whole-wave uniform exit
  const int row0 = (wave / grpN) << 4;
  const int ct0 = (wave % grpN) << 2;
  v8f acc[4] = {};
  wmma4_f32A<KC>(A, lda, row0, Wp, ct0, acc);
  const int lane15 = threadIdx.x & 15;
#pragma unroll
  for (int q = 0; q < 4; ++q) {
    const float bv = bias[((ct0 + q) << 4) + lane15];
#pragma unroll
    for (int r = 0; r < 8; ++r) acc[q][r] += bv;
    st_tile(Out, ldo, row0, (ct0 + q) << 4, acc[q]);
  }
}

// ---------------------------------------------------------------------------
__global__ void init_state(const float* __restrict__ enc_init,
                           const float* __restrict__ ctrl_init,
                           float* __restrict__ hf, float* __restrict__ hb,
                           float* __restrict__ hc, float* __restrict__ cal) {
  const int i = blockIdx.x * blockDim.x + threadIdx.x;
  if (i < BB * EE) {
    const int j = i & (EE - 1);
    hf[i] = enc_init[j];
    hb[i] = enc_init[EE + j];
    hc[i] = ctrl_init[j];
    cal[i] = 0.0f;
  }
}

// ---------------------------------------------------------------------------
// Phase 2: bidirectional GRU scan. blockIdx.x = 0 (fwd) / 1 (bwd).
// h staged in LDS as f16 (WMMA A-operand); f32 master copy in global.
// ---------------------------------------------------------------------------
__global__ void __launch_bounds__(512)
encoder_scan(const float* __restrict__ Gf, const float* __restrict__ Gb,
             const _Float16* __restrict__ Phhf, const float* __restrict__ bhhf,
             const _Float16* __restrict__ Phhb, const float* __restrict__ bhhb,
             float* __restrict__ hf, float* __restrict__ hb,
             float* __restrict__ ghf, float* __restrict__ ghb,
             float* __restrict__ gru_out) {
  __shared__ __align__(32) _Float16 h16s[BB * EE];   // 64 KB
  const int dir = blockIdx.x;
  const float* G = dir ? Gb : Gf;
  const _Float16* Phh = dir ? Phhb : Phhf;
  const float* bhh = dir ? bhhb : bhhf;
  float* h  = dir ? hb  : hf;
  float* gh = dir ? ghb : ghf;
  const int wave   = threadIdx.x >> 5;
  const int nwaves = blockDim.x >> 5;
  const int lane   = threadIdx.x & 31;
  const int UNITS  = (BB / 16) * (G3E / 64);   // 8 x 12 = 96

  for (int e = threadIdx.x; e < BB * EE; e += blockDim.x)
    h16s[e] = (_Float16)h[e];
  __syncthreads();

  for (int s = 0; s < TT; ++s) {
    const int t = dir ? (TT - 1 - s) : s;
    const float* gx = G + (size_t)t * BB * G3E;

    // gh = h @ Whh^T + bhh  (A from LDS f16, B packed f16, 4 tiles per unit)
    for (int unit = wave; unit < UNITS; unit += nwaves) {
      const int row0 = (unit / 12) << 4;
      const int ct0 = (unit % 12) << 2;
      v8f acc[4] = {};
      wmma4_h16A<8>(h16s, EE, row0, Phh, ct0, acc);
#pragma unroll
      for (int q = 0; q < 4; ++q) {
        const float bv = bhh[((ct0 + q) << 4) + (lane & 15)];
#pragma unroll
        for (int r = 0; r < 8; ++r) acc[q][r] += bv;
        st_tile(gh, G3E, row0, (ct0 + q) << 4, acc[q]);
      }
    }
    __threadfence();
    __syncthreads();

    // gates + hidden update (+ refresh LDS f16 copy) + clipped output write
    for (int e = threadIdx.x; e < BB * EE; e += blockDim.x) {
      const int b = e >> 8, j = e & 255;
      const size_t base = (size_t)b * G3E;
      const float r = sigmoidf_(gx[base + j]       + gh[base + j]);
      const float z = sigmoidf_(gx[base + 256 + j] + gh[base + 256 + j]);
      const float n = tanhf(gx[base + 512 + j] + r * gh[base + 512 + j]);
      const float hn = (1.0f - z) * n + z * h[e];
      h[e] = hn;
      h16s[e] = (_Float16)hn;
      const float hcp = fminf(fmaxf(hn, -CLIPV), CLIPV);
      gru_out[((size_t)t * BB + b) * (2 * EE) + (dir ? EE : 0) + j] = hcp;
    }
    // warm the cache with the next step's gx slab
    if (s + 1 < TT)
      prefetch_slab(G + (size_t)(dir ? t - 1 : t + 1) * BB * G3E, BB * G3E);
    __threadfence();
    __syncthreads();
  }
}

// ---------------------------------------------------------------------------
// Phase 4: controller scan (single workgroup). h_c and gi staged in LDS f16.
// ---------------------------------------------------------------------------
__global__ void __launch_bounds__(512)
controller_scan(const float* __restrict__ enc_gates,
                const float* __restrict__ factors,
                const float* __restrict__ eps,
                const _Float16* __restrict__ Pfb,
                const _Float16* __restrict__ Phhc, const float* __restrict__ bhhc,
                const _Float16* __restrict__ Pum,  const float* __restrict__ bum,
                const _Float16* __restrict__ Pul,  const float* __restrict__ bul,
                const _Float16* __restrict__ Psp,  const float* __restrict__ bsp,
                float* __restrict__ hc, float* __restrict__ cal,
                float* __restrict__ S1, float* __restrict__ S2,
                float* __restrict__ out) {
  __shared__ __align__(32) _Float16 hc16s[BB * CC];   // 64 KB
  __shared__ __align__(32) _Float16 gi16s[BB * GG];   // 48 KB
  const int wave   = threadIdx.x >> 5;
  const int nwaves = blockDim.x >> 5;
  const int lane   = threadIdx.x & 31;

  for (int e = threadIdx.x; e < BB * CC; e += blockDim.x)
    hc16s[e] = (_Float16)hc[e];
  __syncthreads();

  for (int t = 0; t < TT; ++t) {
    const float* eg = enc_gates + (size_t)t * BB * G3E;

    // S1 = cal @ Wfb^T ; S2 = hc @ Whhc^T + bhhc   (2 x 96 units of 4 tiles)
    for (int unit = wave; unit < 192; unit += nwaves) {
      const int second = (unit >= 96);
      const int ul = second ? unit - 96 : unit;
      const int row0 = (ul / 12) << 4;
      const int ct0 = (ul % 12) << 2;
      v8f acc[4] = {};
      if (!second) {
        wmma4_f32A<8>(cal, NN, row0, Pfb, ct0, acc);
#pragma unroll
        for (int q = 0; q < 4; ++q)
          st_tile(S1, G3E, row0, (ct0 + q) << 4, acc[q]);
      } else {
        wmma4_h16A<8>(hc16s, CC, row0, Phhc, ct0, acc);
#pragma unroll
        for (int q = 0; q < 4; ++q) {
          const float bv = bhhc[((ct0 + q) << 4) + (lane & 15)];
#pragma unroll
          for (int r = 0; r < 8; ++r) acc[q][r] += bv;
          st_tile(S2, G3E, row0, (ct0 + q) << 4, acc[q]);
        }
      }
    }
    __threadfence();
    __syncthreads();

    // controller GRU update, clipped to +-5 (refresh f32 + LDS f16 copies)
    for (int e = threadIdx.x; e < BB * CC; e += blockDim.x) {
      const int b = e >> 8, j = e & 255;
      const size_t base = (size_t)b * G3E;
      const float r = sigmoidf_(eg[base + j]       + S1[base + j]       + S2[base + j]);
      const float z = sigmoidf_(eg[base + 256 + j] + S1[base + 256 + j] + S2[base + 256 + j]);
      const float n = tanhf(eg[base + 512 + j] + S1[base + 512 + j] + r * S2[base + 512 + j]);
      float hn = (1.0f - z) * n + z * hc[e];
      hn = fminf(fmaxf(hn, -CLIPV), CLIPV);
      hc[e] = hn;
      hc16s[e] = (_Float16)hn;
    }
    // warm the cache with the next step's enc_gates slab
    if (t + 1 < TT)
      prefetch_slab(enc_gates + (size_t)(t + 1) * BB * G3E, BB * G3E);
    __threadfence();
    __syncthreads();

    // u = exp(0.5*(hc@Wul^T+bul))*eps + (hc@Wum^T+bum) ; gi = [u, factors]
    for (int unit = wave; unit < 16; unit += nwaves) { // 8 row-tiles x 2 groups
      const int row0 = (unit >> 1) << 4;
      const int ct0 = (unit & 1) << 2;
      v8f am[4] = {}, al[4] = {};
      wmma4_h16A<8>(hc16s, CC, row0, Pum, ct0, am);
      wmma4_h16A<8>(hc16s, CC, row0, Pul, ct0, al);
      const int hi2 = lane >> 4;
#pragma unroll
      for (int q = 0; q < 4; ++q) {
        const int n = ((ct0 + q) << 4) + (lane & 15);
        const float bm = bum[n], bl = bul[n];
#pragma unroll
        for (int r = 0; r < 8; ++r) {
          const int row = row0 + 8 * hi2 + r;
          const float u = __expf(0.5f * (al[q][r] + bl)) *
                          eps[((size_t)t * BB + row) * UU + n] + (am[q][r] + bm);
          gi16s[row * GG + n] = (_Float16)u;
        }
      }
    }
    for (int e = threadIdx.x; e < BB * FF; e += blockDim.x) {
      const int b = e >> 6, j = e & 63;
      gi16s[b * GG + UU + j] = (_Float16)factors[((size_t)t * BB + b) * FF + j];
    }
    __threadfence();
    __syncthreads();

    // spikes = max(exp(gi@Wsp^T + bsp) - 1, 0); cal = 0.9*cal + spikes; emit
    for (int unit = wave; unit < 32; unit += nwaves) { // 8 row-tiles x 4 groups
      const int row0 = (unit >> 2) << 4;
      const int ct0 = (unit & 3) << 2;
      v8f acc[4] = {};
      wmma4_h16A<6>(gi16s, GG, row0, Psp, ct0, acc);
      const int hi2 = lane >> 4;
#pragma unroll
      for (int q = 0; q < 4; ++q) {
        const int n = ((ct0 + q) << 4) + (lane & 15);
        const float bv = bsp[n];
#pragma unroll
        for (int r = 0; r < 8; ++r) {
          const int row = row0 + 8 * hi2 + r;
          float sp = __expf(acc[q][r] + bv) - 1.0f;
          sp = sp > 0.0f ? sp : 0.0f;
          const size_t ci = (size_t)row * NN + n;
          const float c = DAMPF * cal[ci] + sp;
          cal[ci] = c;
          out[((size_t)t * BB + row) * NN + n] = c;
        }
      }
    }
    __threadfence();
    __syncthreads();
  }
}

// ---------------------------------------------------------------------------
extern "C" void kernel_launch(void* const* d_in, const int* in_sizes, int n_in,
                              void* d_out, int out_size, void* d_ws, size_t ws_size,
                              hipStream_t stream) {
  const float* x        = (const float*)d_in[0];
  const float* factors  = (const float*)d_in[1];
  const float* eps      = (const float*)d_in[2];
  const float* Wihf     = (const float*)d_in[3];
  const float* Whhf     = (const float*)d_in[4];
  const float* bihf     = (const float*)d_in[5];
  const float* bhhf     = (const float*)d_in[6];
  const float* Wihb     = (const float*)d_in[7];
  const float* Whhb     = (const float*)d_in[8];
  const float* bihb     = (const float*)d_in[9];
  const float* bhhb     = (const float*)d_in[10];
  const float* enc_init = (const float*)d_in[11];
  const float* ctrl_init= (const float*)d_in[12];
  const float* Wihc     = (const float*)d_in[13];
  const float* Whhc     = (const float*)d_in[14];
  const float* bihc     = (const float*)d_in[15];
  const float* bhhc     = (const float*)d_in[16];
  const float* Wum      = (const float*)d_in[17];
  const float* bum      = (const float*)d_in[18];
  const float* Wul      = (const float*)d_in[19];
  const float* bul      = (const float*)d_in[20];
  const float* Wsp      = (const float*)d_in[21];
  const float* bsp      = (const float*)d_in[22];
  float* out = (float*)d_out;

  float* ws = (float*)d_ws;
  size_t off = 0;
  auto allocf = [&](size_t n) { float* p = ws + off; off += (n + 15) & ~(size_t)15; return p; };

  float* Gf   = allocf((size_t)TT * BB * G3E);
  float* Gb   = allocf((size_t)TT * BB * G3E);
  float* gru  = allocf((size_t)TT * BB * (2 * EE));
  float* egts = allocf((size_t)TT * BB * G3E);
  float* hf   = allocf((size_t)BB * EE);
  float* hb   = allocf((size_t)BB * EE);
  float* ghf  = allocf((size_t)BB * G3E);
  float* ghb  = allocf((size_t)BB * G3E);
  float* hc   = allocf((size_t)BB * CC);
  float* cal  = allocf((size_t)BB * NN);
  float* S1   = allocf((size_t)BB * G3E);
  float* S2   = allocf((size_t)BB * G3E);
  // packed f16 weights (sizes in halves / 2 = floats)
  _Float16* Pihf = (_Float16*)allocf((size_t)G3E * NN / 2);
  _Float16* Pihb = (_Float16*)allocf((size_t)G3E * NN / 2);
  _Float16* Penc = (_Float16*)allocf((size_t)G3E * 512 / 2);
  _Float16* Pfb  = (_Float16*)allocf((size_t)G3E * NN / 2);
  _Float16* Phhf = (_Float16*)allocf((size_t)G3E * EE / 2);
  _Float16* Phhb = (_Float16*)allocf((size_t)G3E * EE / 2);
  _Float16* Phhc = (_Float16*)allocf((size_t)G3E * CC / 2);
  _Float16* Pum  = (_Float16*)allocf((size_t)UU * CC / 2);
  _Float16* Pul  = (_Float16*)allocf((size_t)UU * CC / 2);
  _Float16* Psp  = (_Float16*)allocf((size_t)NN * GG / 2);

  // --- one-time weight packing (f32 -> f16 B-fragment layout) ---
  auto pk = [&](const float* W, int ldw, int co, int K, int Nc, _Float16* dst) {
    pack_weight<<<(Nc * K + 511) / 512, 512, 0, stream>>>(W, ldw, co, K, Nc, dst);
  };
  pk(Wihf, NN, 0, NN, G3E, Pihf);
  pk(Wihb, NN, 0, NN, G3E, Pihb);
  pk(Wihc, G3E, 0, 512, G3E, Penc);
  pk(Wihc, G3E, 512, NN, G3E, Pfb);
  pk(Whhf, EE, 0, EE, G3E, Phhf);
  pk(Whhb, EE, 0, EE, G3E, Phhb);
  pk(Whhc, CC, 0, CC, G3E, Phhc);
  pk(Wum, CC, 0, CC, UU, Pum);
  pk(Wul, CC, 0, CC, UU, Pul);
  pk(Wsp, GG, 0, GG, NN, Psp);

  init_state<<<(BB * EE + 255) / 256, 256, 0, stream>>>(enc_init, ctrl_init,
                                                        hf, hb, hc, cal);

  {
    const int M = TT * BB;
    const int units = (M / 16) * (G3E / 64);       // 16x64 strips
    const int blocks = (units + 7) / 8;            // 8 waves / block
    gemm_wmma<8><<<blocks, 256, 0, stream>>>(x, NN, Pihf, bihf, Gf, G3E, M, G3E);
    gemm_wmma<8><<<blocks, 256, 0, stream>>>(x, NN, Pihb, bihb, Gb, G3E, M, G3E);
  }

  encoder_scan<<<2, 512, 0, stream>>>(Gf, Gb, Phhf, bhhf, Phhb, bhhb,
                                      hf, hb, ghf, ghb, gru);

  {
    const int M = TT * BB;
    const int units = (M / 16) * (G3E / 64);
    const int blocks = (units + 7) / 8;
    gemm_wmma<16><<<blocks, 256, 0, stream>>>(gru, 2 * EE, Penc, bihc,
                                              egts, G3E, M, G3E);
  }

  controller_scan<<<1, 512, 0, stream>>>(egts, factors, eps, Pfb, Phhc, bhhc,
                                         Pum, bum, Pul, bul, Psp, bsp,
                                         hc, cal, S1, S2, out);
}
